// Predictor_75917841924358
// MI455X (gfx1250) — compile-verified, matrix operand order checked by reading, not run on hardware
//
#include <hip/hip_runtime.h>
#include <hip/hip_bf16.h>
#include <math.h>

// ---------------------------------------------------------------------------
// Types for CDNA5 WMMA
// ---------------------------------------------------------------------------
typedef __attribute__((ext_vector_type(16))) __bf16 v16bf;
typedef __attribute__((ext_vector_type(8)))  float  v8f;

__device__ __forceinline__ unsigned short f32_to_bf16(float f) {
  unsigned int u = __float_as_uint(f);
  u += 0x7FFFu + ((u >> 16) & 1u);   // round to nearest even
  return (unsigned short)(u >> 16);
}

// ---------------------------------------------------------------------------
// Implicit-GEMM 3x3 conv (stride 1, SAME) + bias + ReLU, bf16 WMMA, f32 acc.
// K ordering: k = (r*3+s)*IC + ic  (weights pre-transposed to [rs][OC][IC]),
// so every 32-wide K-step sits inside one (r,s) tap: no div/mod or bounds
// recomputation in the ic inner loop.  Input is always packed bf16.
//   in : [B][IC][H][W]   bf16 (pre-converted / pre-gathered)
//   wq : [9][OC][IC]     bf16
//   out: [B][OC][H][W]   (bf16 or f32, OUT_BF16)
// Block: 256 threads = 8 waves (2 wm x 4 wn), each wave 2x2 16x16 tiles
//   -> block tile 64 OC x 128 pixels, 4 WMMA / wave / K-step, LDS ping-pong.
// ---------------------------------------------------------------------------
template <bool OUT_BF16>
__global__ __launch_bounds__(256)
void conv3x3_wmma(const unsigned short* __restrict__ in_h,
                  const unsigned short* __restrict__ wq,
                  const float* __restrict__ bias,
                  void* __restrict__ outv,
                  int IC, int OC, int H, int Wd)
{
  const int nPix = H * Wd;
  const int b    = blockIdx.z;
  const long inBase = (long)b * IC * nPix;
  const int pix0 = blockIdx.x * 128;
  const int oc0  = blockIdx.y * 64;

  __shared__ unsigned short As[2][64][36];    // [buf][m][k] (+pad, rows 72B)
  __shared__ unsigned short Bs[2][128][36];   // [buf][n][k]

  const int tid  = threadIdx.x;
  const int wave = tid >> 5, lane = tid & 31;
  const int wm = wave >> 2, wn = wave & 3;
  const int lg = lane >> 4, lr = lane & 15;

  // B staging: one fixed pixel per thread, 16 k-slots (kk = kb0 + 2q)
  const int  bn  = tid & 127;
  const int  kb0 = tid >> 7;
  const int  bp  = pix0 + bn;
  const bool pvalid = bp < nPix;
  const int  py = pvalid ? bp / Wd : 0;
  const int  px = bp - py * Wd;

  v8f acc[2][2] = {};

  const int nKs   = IC >> 5;       // ic-steps per (r,s) tap
  const int steps = 9 * nKs;

  auto stage = [&](int r, int s, int ic0, int bufi) {
    // ---- A: 64 oc x 32 ic = 1024 aligned pair loads, 4/thread (b128-able)
    const long wrow = (long)((r * 3 + s) * OC + oc0) * IC + ic0;
#pragma unroll
    for (int q = 0; q < 4; ++q) {
      int p  = tid * 4 + q;             // pair id 0..1023
      int m  = p >> 4;
      int kk = (p & 15) << 1;
      *(unsigned int*)&As[bufi][m][kk] =
          *(const unsigned int*)&wq[wrow + (long)m * IC + kk];
    }
    // ---- B: 128 pix x 32 ic, 16/thread; tap offset fixed per (r,s)
    int  iy = py + r - 1, ix = px + s - 1;
    bool valid = pvalid && iy >= 0 && iy < H && ix >= 0 && ix < Wd;
    long base  = inBase + (long)iy * Wd + ix + (long)ic0 * nPix;
#pragma unroll
    for (int q = 0; q < 16; ++q) {
      int kk = kb0 + (q << 1);
      unsigned short val = 0;
      if (valid) val = in_h[base + (long)kk * nPix];
      Bs[bufi][bn][kk] = val;
    }
  };

  auto compute = [&](int bufi) {
    union { unsigned int u[8]; v16bf v; } fa[2], fb[2];
#pragma unroll
    for (int t2 = 0; t2 < 2; ++t2) {
#pragma unroll
      for (int v = 0; v < 8; ++v) {
        int kb = (v < 4) ? (2 * v + 8 * lg) : (16 + 2 * (v - 4) + 8 * lg);
        fa[t2].u[v] = *(const unsigned int*)&As[bufi][wm * 32 + t2 * 16 + lr][kb];
        fb[t2].u[v] = *(const unsigned int*)&Bs[bufi][wn * 32 + t2 * 16 + lr][kb];
      }
    }
#ifdef __AMDGCN__
#pragma unroll
    for (int mt = 0; mt < 2; ++mt)
#pragma unroll
      for (int nt = 0; nt < 2; ++nt)
        acc[mt][nt] = __builtin_amdgcn_wmma_f32_16x16x32_bf16(
            false, fa[mt].v, false, fb[nt].v, (short)0, acc[mt][nt], false, false);
#endif
  };

  // software pipeline: always stage st+1 while computing st; peel last compute
  stage(0, 0, 0, 0);
  int r_c = 0, s_c = 0, ic_c = 32;
  if (ic_c == IC) { ic_c = 0; s_c = 1; }
  __syncthreads();

  int buf = 0;
  for (int st = 0; st < steps - 1; ++st) {
    stage(r_c, s_c, ic_c, buf ^ 1);
    ic_c += 32;
    if (ic_c == IC) {
      ic_c = 0;
      if (++s_c == 3) { s_c = 0; ++r_c; }
    }
    compute(buf);
    __syncthreads();
    buf ^= 1;
  }
  compute(buf);

  // epilogue: C/D layout -> M = v + 8*lg, N = lr
#pragma unroll
  for (int mt = 0; mt < 2; ++mt) {
#pragma unroll
    for (int nt = 0; nt < 2; ++nt) {
#pragma unroll
      for (int v = 0; v < 8; ++v) {
        int m  = v + 8 * lg;
        int oc = oc0 + wm * 32 + mt * 16 + m;
        int p  = pix0 + wn * 32 + nt * 16 + lr;
        if (p < nPix) {
          float rv = acc[mt][nt][v] + bias[oc];
          rv = rv > 0.f ? rv : 0.f;
          long oidx = (long)b * OC * nPix + (long)oc * nPix + p;
          if (OUT_BF16) ((unsigned short*)outv)[oidx] = f32_to_bf16(rv);
          else          ((float*)outv)[oidx] = rv;
        }
      }
    }
  }
}

// ---------------------------------------------------------------------------
// Weight conversion f32 OIHW -> bf16 [rs][OC][IC]
// ---------------------------------------------------------------------------
__global__ void cvt_w_rs(const float* __restrict__ src,
                         unsigned short* __restrict__ dst, int OC, int IC) {
  const long n = (long)OC * IC * 9;
  const long ocic = (long)OC * IC;
  long i = (long)blockIdx.x * blockDim.x + threadIdx.x;
  long stride = (long)gridDim.x * blockDim.x;
  for (; i < n; i += stride) {
    long rs  = i / ocic;
    long rem = i - rs * ocic;
    long oc  = rem / IC;
    long ic  = rem - oc * IC;
    dst[i] = f32_to_bf16(src[(oc * IC + ic) * 9 + rs]);
  }
}

// plain elementwise f32 -> bf16
__global__ void cvt_f32_bf16(const float* __restrict__ src,
                             unsigned short* __restrict__ dst, int n) {
  int i = blockIdx.x * blockDim.x + threadIdx.x;
  int stride = gridDim.x * blockDim.x;
  for (; i < n; i += stride) dst[i] = f32_to_bf16(src[i]);
}

// gather the 10 selected ROI boxes and convert to packed bf16 [10][1024][196]
__global__ void gather_roi_bf16(const float* __restrict__ roi,
                                const int* __restrict__ sel,
                                unsigned short* __restrict__ dst) {
  const long boxElems = 1024L * 196L;
  int bx = blockIdx.y;                       // 0..9
  long src0 = (long)sel[bx] * boxElems;
  long dst0 = (long)bx * boxElems;
  for (long i = (long)blockIdx.x * blockDim.x + threadIdx.x; i < boxElems;
       i += (long)gridDim.x * blockDim.x)
    dst[dst0 + i] = f32_to_bf16(roi[src0 + i]);
}

// ---------------------------------------------------------------------------
// Scoring / selection / heads
// ---------------------------------------------------------------------------
__global__ void cm_kernel(const float* __restrict__ g2, float* __restrict__ cm) {
  int p = blockIdx.x * blockDim.x + threadIdx.x;
  if (p < 3600) {
    float s = 0.f;
    for (int c = 0; c < 256; ++c) s += g2[c * 3600 + p];
    cm[p] = s * (1.f / 256.f);
  }
}

__global__ void globp_kernel(const float* __restrict__ g2, float* __restrict__ gp) {
  int c = threadIdx.x;
  float s = 0.f;
  const float* src = g2 + (long)c * 3600;
  for (int p = 0; p < 3600; ++p) s += src[p];
  gp[c] = s * (1.f / 3600.f);
}

__global__ void select_kernel(const float* __restrict__ cm,
                              const int* __restrict__ bbox,
                              int* __restrict__ sel, float* __restrict__ gsel) {
  const int N = 300, K = 10;
  __shared__ float sc[300];
  __shared__ float gated[300];
  __shared__ unsigned char picked[300];
  __shared__ float thresh;
  int n = threadIdx.x;
  if (n < N) {
    int x1 = bbox[n * 4 + 0] >> 4; if (x1 > 79) x1 = 79;
    int y1 = bbox[n * 4 + 1] >> 4; if (y1 > 44) y1 = 44;
    int x2 = bbox[n * 4 + 2] >> 4; if (x2 > 79) x2 = 79;
    int y2 = bbox[n * 4 + 3] >> 4; if (y2 > 44) y2 = 44;
    float s = 0.f;
    for (int yy = y1; yy <= y2; ++yy)
      for (int xx = x1; xx <= x2; ++xx) s += cm[yy * 80 + xx];
    sc[n] = s / (float)((y2 - y1 + 1) * (x2 - x1 + 1));
    picked[n] = 0;
  }
  __syncthreads();
  if (n == 0) {
    float s9 = 0.f, s10 = 0.f;
    for (int j = 0; j < K + 1; ++j) {
      int best = 0; float bv = -1e30f;
      for (int m = 0; m < N; ++m)
        if (!picked[m] && sc[m] > bv) { bv = sc[m]; best = m; }
      picked[best] = 1;
      if (j == K - 1) s9 = bv;
      if (j == K)     s10 = bv;
    }
    thresh = 0.5f * (s9 + s10);
  }
  __syncthreads();
  if (n < N) gated[n] = 1.f / (1.f + expf(-(sc[n] - thresh) * 100.f));
  __syncthreads();
  if (n == 0) {
    int tmp[10]; int cnt = 0;
    for (int m = 0; m < N && cnt < 10; ++m)
      if (gated[m] > 0.49f) tmp[cnt++] = m;
    if (cnt == 0) { tmp[0] = 0; cnt = 1; }
    for (int i = 0; i < K; ++i) {
      int s = tmp[i % cnt];
      sel[i] = s;
      gsel[i] = gated[s];
    }
  }
}

__global__ void heads1_kernel(const float* __restrict__ r2,
                              const float* __restrict__ gsel,
                              const float* __restrict__ gp,
                              const float* __restrict__ Wf1,
                              const float* __restrict__ bf1,
                              float* __restrict__ h) {
  __shared__ float feat[512];
  int i = blockIdx.x, c = threadIdx.x;
  const float* src = r2 + ((long)i * 256 + c) * 196;
  float s = 0.f;
  for (int j = 0; j < 196; ++j) s += src[j];
  feat[c] = gsel[i] * s * (1.f / 196.f);
  feat[256 + c] = gp[c];
  __syncthreads();
  float a = bf1[c];
  const float* w = Wf1 + (long)c * 512;
  for (int j = 0; j < 512; ++j) a += feat[j] * w[j];
  h[i * 256 + c] = a > 0.f ? a : 0.f;
}

__global__ void heads2_kernel(const float* __restrict__ h,
                              const float* __restrict__ Wf2,
                              const float* __restrict__ bf2,
                              const float* __restrict__ Wf3,
                              const float* __restrict__ bf3,
                              float* __restrict__ out) {
  __shared__ float h2s[64];
  int c = threadIdx.x;
  if (c < 64) {
    float a = bf2[c];
    const float* w = Wf2 + (long)c * 2560;
    for (int j = 0; j < 2560; ++j) a += h[j] * w[j];
    h2s[c] = a > 0.f ? a : 0.f;
  }
  __syncthreads();
  if (c < 4) {
    float a = bf3[c];
    const float* w = Wf3 + (long)c * 64;
    for (int j = 0; j < 64; ++j) a += h2s[j] * w[j];
    out[c] = a;
  }
}

// ---------------------------------------------------------------------------
// Host launcher
// ---------------------------------------------------------------------------
extern "C" void kernel_launch(void* const* d_in, const int* in_sizes, int n_in,
                              void* d_out, int out_size, void* d_ws, size_t ws_size,
                              hipStream_t stream) {
  (void)in_sizes; (void)n_in; (void)out_size; (void)ws_size;

  const float* glob = (const float*)d_in[0];   // [1,1024,45,80]
  const float* roi  = (const float*)d_in[1];   // [300,1024,14,14]
  const int*   bbox = (const int*)d_in[2];     // [300,4]
  const float* Wc1  = (const float*)d_in[3];   // [512,1024,3,3]
  const float* bc1  = (const float*)d_in[4];
  const float* Wc2  = (const float*)d_in[5];   // [256,512,3,3]
  const float* bc2  = (const float*)d_in[6];
  const float* Wf1  = (const float*)d_in[7];   // [256,512]
  const float* bf1  = (const float*)d_in[8];
  const float* Wf2  = (const float*)d_in[9];   // [64,2560]
  const float* bf2  = (const float*)d_in[10];
  const float* Wf3  = (const float*)d_in[11];  // [4,64]
  const float* bf3  = (const float*)d_in[12];
  float* out = (float*)d_out;

  char* ws = (char*)d_ws;
  size_t off = 0;
  auto walloc = [&](size_t bytes) -> void* {
    void* p = ws + off;
    off += (bytes + 255) & ~(size_t)255;
    return p;
  };
  unsigned short* wq1 = (unsigned short*)walloc((size_t)512 * 9216 * 2);  // bf16 [9][512][1024]
  unsigned short* wq2 = (unsigned short*)walloc((size_t)256 * 4608 * 2);  // bf16 [9][256][512]
  unsigned short* gin = (unsigned short*)walloc((size_t)1024 * 3600 * 2); // glob input (bf16)
  unsigned short* g1  = (unsigned short*)walloc((size_t)512 * 3600 * 2);  // glob conv1 (bf16)
  float*          g2  = (float*)walloc((size_t)256 * 3600 * 4);           // glob conv2 (f32)
  float*          cm  = (float*)walloc((size_t)3600 * 4);
  float*          gp  = (float*)walloc((size_t)256 * 4);
  int*            sel = (int*)walloc((size_t)10 * 4);
  float*          gsl = (float*)walloc((size_t)10 * 4);
  unsigned short* rin = (unsigned short*)walloc((size_t)10 * 1024 * 196 * 2); // sel ROI (bf16)
  unsigned short* r1  = (unsigned short*)walloc((size_t)10 * 512 * 196 * 2);  // roi conv1 (bf16)
  float*          r2  = (float*)walloc((size_t)10 * 256 * 196 * 4);           // roi conv2 (f32)
  float*          h   = (float*)walloc((size_t)2560 * 4);

  // weights -> bf16 [rs][OC][IC]; glob input -> bf16
  cvt_w_rs<<<1024, 256, 0, stream>>>(Wc1, wq1, 512, 1024);
  cvt_w_rs<<<512, 256, 0, stream>>>(Wc2, wq2, 256, 512);
  cvt_f32_bf16<<<2048, 256, 0, stream>>>(glob, gin, 1024 * 3600);

  // global path: conv1 (bf16 out) -> conv2 (f32 out); pixel tiles of 128
  conv3x3_wmma<true><<<dim3(29, 8, 1), 256, 0, stream>>>(
      gin, wq1, bc1, g1, 1024, 512, 45, 80);
  conv3x3_wmma<false><<<dim3(29, 4, 1), 256, 0, stream>>>(
      g1, wq2, bc2, g2, 512, 256, 45, 80);

  // scoring + selection (depends only on the global path)
  cm_kernel<<<15, 256, 0, stream>>>(g2, cm);
  globp_kernel<<<1, 256, 0, stream>>>(g2, gp);
  select_kernel<<<1, 320, 0, stream>>>(cm, bbox, sel, gsl);

  // gather + convert only the 10 selected boxes, then run their convs
  gather_roi_bf16<<<dim3(128, 10), 256, 0, stream>>>(roi, sel, rin);
  conv3x3_wmma<true><<<dim3(2, 8, 10), 256, 0, stream>>>(
      rin, wq1, bc1, r1, 1024, 512, 14, 14);
  conv3x3_wmma<false><<<dim3(2, 4, 10), 256, 0, stream>>>(
      r1, wq2, bc2, r2, 512, 256, 14, 14);

  // FC heads
  heads1_kernel<<<10, 256, 0, stream>>>(r2, gsl, gp, Wf1, bf1, h);
  heads2_kernel<<<1, 64, 0, stream>>>(h, Wf2, bf2, Wf3, bf3, out);
}
